// static_graph_23192823399236
// MI455X (gfx1250) — matrix operand order repaired and checked by reading, compiled, not run on hardware
//
#include <hip/hip_runtime.h>
#include <hip/hip_fp16.h>

#define NT 100000
#define NB 50
#define NE 1600000
#define D 128
#define NPG (NT / NB)   // 2000 nodes per graph
#define LDH 136         // padded LDS row stride (halfs) to break bank conflicts

typedef __attribute__((ext_vector_type(16))) _Float16 v16h;
typedef __attribute__((ext_vector_type(8)))  float    v8f;

// ---------------- utility: zero a float buffer (grid-stride) ----------------
__global__ void zero_f32(float* __restrict__ p, long n) {
    long i = (long)blockIdx.x * blockDim.x + threadIdx.x;
    long stride = (long)gridDim.x * blockDim.x;
    for (; i < n; i += stride) p[i] = 0.0f;
}

// ---- one-time prep: W (fp32, k-major [k][n]) -> W^T f16 (n-major [n][k]) ----
__global__ void prep_weights(const float* __restrict__ W1, const float* __restrict__ W2,
                             __half* __restrict__ W1t, __half* __restrict__ W2t) {
    int i = blockIdx.x * 256 + threadIdx.x;
    if (i < D * D) {
        int n = i >> 7, k = i & (D - 1);
        W1t[n * D + k] = __float2half(W1[k * D + n]);
        W2t[n * D + k] = __float2half(W2[k * D + n]);
    }
}

// ---------------- in-degree via f32 atomics (L2-resident) ----------------
__global__ void deg_kernel(const int* __restrict__ dst, float* __restrict__ deg) {
    int e = blockIdx.x * 256 + threadIdx.x;
    if (e < NE) atomicAdd(&deg[dst[e]], 1.0f);
}

// ------------- WMMA projection: Hout = gatherOrH @ W + b  -------------
// block = 256 threads (8 waves, wave32). blockIdx.x -> 16 rows.
// wave w computes the 16x16 tile covering output columns [16w, 16w+16).
__global__ __launch_bounds__(256) void proj_wmma(
    const float* __restrict__ Hin,       // [NT,D] fp32 (layer-2 path)
    const int*   __restrict__ node_ids,  // layer-1 gather indices
    const float* __restrict__ embeds,    // [V,D]
    const __half* __restrict__ Wt,       // [D][D] f16, n-major (W transposed)
    const float* __restrict__ bias,      // [D]
    float* __restrict__ Hout,            // [NT,D] fp32
    int gather)
{
    __shared__ __half sH[16 * LDH];   // 16 activation rows, f16
    __shared__ __half sW[D * LDH];    // full W^T, f16

    const int t = threadIdx.x;
    const int row0 = blockIdx.x * 16;

    // stage activations (fp32 -> f16), gathered through node_ids for layer 1
    for (int idx = t; idx < 16 * D; idx += 256) {
        int r = idx >> 7, c = idx & (D - 1);
        int g = row0 + r;
        float v = gather ? embeds[(long)node_ids[g] * D + c]
                         : Hin[(long)g * D + c];
        sH[r * LDH + c] = __float2half(v);
    }
    // stage W^T (already f16 in global)
    for (int idx = t; idx < D * D; idx += 256) {
        int n = idx >> 7, k = idx & (D - 1);
        sW[n * LDH + k] = Wt[idx];
    }
    __syncthreads();

    const int wave = t >> 5;
    const int lane = t & 31;
    const int hi   = lane >> 4;       // half-wave select
    const int mn   = lane & 15;       // row (A) / col (B) index within tile

    const __half* ha = &sH[mn * LDH];                // A: row mn of activations
    const __half* hb = &sW[(wave * 16 + mn) * LDH];  // B: output column wave*16+mn

    union Frag { v16h v; unsigned u[8]; };
    v8f acc = {};

    #pragma unroll
    for (int kb = 0; kb < 4; ++kb) {                 // K = 128 in 4 x 32
        Frag A, Bf;
        #pragma unroll
        for (int j = 0; j < 8; ++j) {
            // A layout (16-bit, 16x32): VGPR0-3 -> K = hi*8 + {0..7},
            //                           VGPR4-7 -> K = 16 + hi*8 + {0..7}
            int ka = kb * 32 + ((j < 4) ? (hi * 8 + 2 * j)
                                        : (16 + hi * 8 + 2 * (j - 4)));
            // B layout (16-bit, 32x16): same K order as A
            int kbb = ka;
            A.u[j]  = *(const unsigned*)(ha + ka);
            Bf.u[j] = *(const unsigned*)(hb + kbb);
        }
        acc = __builtin_amdgcn_wmma_f32_16x16x32_f16(
                  false, A.v, false, Bf.v, (short)0, acc, false, false);
    }

    const int ncol = wave * 16 + mn;
    const float bv = bias[ncol];
    #pragma unroll
    for (int r = 0; r < 8; ++r) {
        int m = hi * 8 + r;                          // C/D layout: lanes16-31 hold M=8..15
        Hout[(long)(row0 + m) * D + ncol] = acc[r] + bv;
    }
}

// ------- edge scatter: S[dst] += Wh[src] * w  (8 edges/block, 32 lanes/edge) -------
__global__ __launch_bounds__(256) void edge_agg(
    const float* __restrict__ Wh, const int* __restrict__ src,
    const int* __restrict__ dst, const float* __restrict__ ew,
    float* __restrict__ S)
{
    int t = threadIdx.x;
    int e = blockIdx.x * 8 + (t >> 5);
    int lane = t & 31;
    if (e >= NE) return;
    int s = src[e], d = dst[e];
    float w = ew[e];
    const float4 v = *(const float4*)&Wh[(long)s * D + lane * 4];
    float* o = &S[(long)d * D + lane * 4];
    atomicAdd(o + 0, v.x * w);
    atomicAdd(o + 1, v.y * w);
    atomicAdd(o + 2, v.z * w);
    atomicAdd(o + 3, v.w * w);
}

// ------- mean over in-edges + leaky_relu(0.01) -------
__global__ void mean_lrelu(const float* __restrict__ S, const float* __restrict__ deg,
                           float* __restrict__ H)
{
    long i = (long)blockIdx.x * 256 + threadIdx.x;
    if (i >= (long)NT * D) return;
    long n = i >> 7;
    float dg = deg[n];
    float v = (dg > 0.0f) ? S[i] / fmaxf(dg, 1.0f) : 0.0f;
    H[i] = (v > 0.0f) ? v : 0.01f * v;
}

// ------- fused layer-2 mean + per-graph mean pool: pool[b][d] -------
__global__ __launch_bounds__(128) void pool_kernel(
    const float* __restrict__ S, const float* __restrict__ deg,
    float* __restrict__ pool)
{
    int b = blockIdx.x, d = threadIdx.x;
    float acc = 0.0f;
    for (int i = 0; i < NPG; ++i) {
        long n = (long)b * NPG + i;
        float dg = deg[n];
        float s = S[n * D + d];
        acc += (dg > 0.0f) ? s / fmaxf(dg, 1.0f) : 0.0f;
    }
    pool[b * D + d] = acc / (float)NPG;
}

// ------- head: logits, sigmoid, stable BCE mean. out = [loss, y_pred x 50] -------
__global__ __launch_bounds__(64) void head_kernel(
    const float* __restrict__ pool, const float* __restrict__ Wout,
    const float* __restrict__ bout, const float* __restrict__ y,
    float* __restrict__ out)
{
    __shared__ float lterm[64];
    int t = threadIdx.x;
    float loss = 0.0f;
    if (t < NB) {
        float acc = 0.0f;
        for (int d2 = 0; d2 < D; ++d2) acc += pool[t * D + d2] * Wout[d2];
        float logit = acc + bout[0];
        out[1 + t] = 1.0f / (1.0f + expf(-logit));
        loss = fmaxf(logit, 0.0f) - logit * y[t] + log1pf(expf(-fabsf(logit)));
    }
    lterm[t] = loss;
    __syncthreads();
    for (int off = 32; off > 0; off >>= 1) {
        if (t < off) lterm[t] += lterm[t + off];
        __syncthreads();
    }
    if (t == 0) out[0] = lterm[0] / (float)NB;
}

extern "C" void kernel_launch(void* const* d_in, const int* in_sizes, int n_in,
                              void* d_out, int out_size, void* d_ws, size_t ws_size,
                              hipStream_t stream) {
    const int*   node_ids = (const int*)d_in[0];
    const int*   src      = (const int*)d_in[1];
    const int*   dst      = (const int*)d_in[2];
    const float* ew       = (const float*)d_in[3];
    const float* y        = (const float*)d_in[4];
    const float* embeds   = (const float*)d_in[5];
    const float* W1       = (const float*)d_in[6];
    const float* b1       = (const float*)d_in[7];
    const float* W2       = (const float*)d_in[8];
    const float* b2       = (const float*)d_in[9];
    const float* Wout     = (const float*)d_in[10];
    const float* bout     = (const float*)d_in[11];
    float* out = (float*)d_out;

    // workspace layout (256B aligned slots)
    char* ws = (char*)d_ws;
    const size_t FEAT = (size_t)NT * D * sizeof(float);        // 51.2 MB
    float*  bufA = (float*)(ws);                               // Wh (both layers)
    float*  bufB = (float*)(ws + FEAT);                        // scatter target s
    float*  bufC = (float*)(ws + 2 * FEAT);                    // h1 (post layer 1)
    size_t  off  = 3 * FEAT;
    float*  deg  = (float*)(ws + off);  off += ((size_t)NT * 4 + 255) & ~255ULL;
    __half* W1t  = (__half*)(ws + off); off += (size_t)D * D * 2;
    __half* W2t  = (__half*)(ws + off); off += (size_t)D * D * 2;
    float*  pool = (float*)(ws + off);

    // prep: f16 transposed weights; zero degree buffer; compute degrees
    prep_weights<<<(D * D + 255) / 256, 256, 0, stream>>>(W1, W2, W1t, W2t);
    zero_f32<<<512, 256, 0, stream>>>(deg, NT);
    deg_kernel<<<(NE + 255) / 256, 256, 0, stream>>>(dst, deg);

    // ---- layer 1 ----
    proj_wmma<<<NT / 16, 256, 0, stream>>>(nullptr, node_ids, embeds, W1t, b1, bufA, 1);
    zero_f32<<<2048, 256, 0, stream>>>(bufB, (long)NT * D);
    edge_agg<<<NE / 8, 256, 0, stream>>>(bufA, src, dst, ew, bufB);
    mean_lrelu<<<(NT * D) / 256, 256, 0, stream>>>(bufB, deg, bufC);

    // ---- layer 2 ----
    proj_wmma<<<NT / 16, 256, 0, stream>>>(bufC, nullptr, nullptr, W2t, b2, bufA, 0);
    zero_f32<<<2048, 256, 0, stream>>>(bufB, (long)NT * D);
    edge_agg<<<NE / 8, 256, 0, stream>>>(bufA, src, dst, ew, bufB);

    // ---- pool + head ----
    pool_kernel<<<NB, 128, 0, stream>>>(bufB, deg, pool);
    head_kernel<<<1, 64, 0, stream>>>(pool, Wout, bout, y, out);
}
